// MHA_OLD_5454608466692
// MI455X (gfx1250) — compile-verified, matrix operand order checked by reading, not run on hardware
//
#include <hip/hip_runtime.h>
#include <hip/hip_bf16.h>

// ---------------------------------------------------------------------------
// MI455X (gfx1250) implementation of the PSTNet-style point-cloud forecaster.
//  - Dense layers run on v_wmma_f32_16x16x32_f16. Each wave32 owns a 16x64
//    output tile (4 f32 accumulators): per 32-wide K-chunk the A fragment is
//    loaded once and ALL 8 B-fragment b128 loads are issued up-front, then 4
//    WMMAs drain them, so load latency overlaps matrix work. Weights are
//    pre-packed ONCE to f16, transposed and zero-padded ([Npad%64][Kpad%32])
//    so every B load is an unconditional aligned b128; the A side has a
//    wave-uniform fast path and a branch-free clamped edge path.
//  - FPS keeps BOTH the xyz cloud (192KB) and the running distance array
//    (64KB) resident in CDNA5's 320KB WGP LDS: the 512 serial iterations
//    never touch global memory after the initial stage-in.
//  - kNN keeps its distance array in LDS and does k argmin rounds.
//  - Neighbor attention (k<=32), temporal MHA (Tm<=2) and inverse-distance
//    interpolation are fused VALU kernels (too small/jagged for WMMA).
// ---------------------------------------------------------------------------

typedef __attribute__((ext_vector_type(16))) _Float16 v16h;
typedef __attribute__((ext_vector_type(8)))  _Float16 v8h;
typedef __attribute__((ext_vector_type(8)))  float    v8f;

#define GRID_STRIDE(t, total) \
  for (size_t t = (size_t)blockIdx.x * blockDim.x + threadIdx.x; t < (total); \
       t += (size_t)gridDim.x * blockDim.x)

// ---------------- weight pre-pack: f32 [K,N] -> f16 transposed [Np][Kp] -----
__global__ void k_packw(const float* __restrict__ W, _Float16* __restrict__ WT,
                        int K, int N, int Kp, int Np)
{
  size_t total = (size_t)Np * Kp;
  GRID_STRIDE(t, total) {
    int k = (int)(t % Kp);
    int n = (int)(t / Kp);
    float v = (k < K && n < N) ? W[(size_t)k * N + n] : 0.f;
    WT[t] = (_Float16)v;
  }
}

// ---------------- WMMA GEMM:  C[M,N] = act(A[M,K] @ W[K,N] + bias) ----------
// WT: packed f16 weight, [Npad][Kpad] (transposed, zero-padded, Npad%64==0).
// blockDim = 256 (8 waves); each wave owns a 16x64 output tile (4 accs).
// A fragment (16x32 f16, ISA 7.12.2): lane<16 holds row M=lane with
// K = {0..7, 16..23}; lane>=16 holds K = {8..15, 24..31}. B mirrors on cols.
// C/D layout: VGPR r, lane l -> (M = r + 8*(l>>4), N = l&15).
__global__ void k_gemm(const float* __restrict__ A, const _Float16* __restrict__ WT,
                       const float* __restrict__ bias, float* __restrict__ C,
                       int M, int K, int N, int Kpad, int relu)
{
  const int ntn4 = (N + 63) >> 6;                 // 64-wide column groups
  const int ntm  = (M + 15) >> 4;
  const int tile = blockIdx.x * (blockDim.x >> 5) + (threadIdx.x >> 5);
  if (tile >= ntn4 * ntm) return;
  const int tm   = (tile / ntn4) << 4;
  const int tn   = (tile % ntn4) << 6;
  const int lane = threadIdx.x & 31;
  const int half = lane >> 4;
  const int l15  = lane & 15;
  const int arow = tm + l15;
  const bool rowok  = (arow < M);
  const bool tileMf = (tm + 16 <= M);             // wave-uniform fast-path flag

  v8f acc[4];
  const _Float16* wrow[4];
#pragma unroll
  for (int j = 0; j < 4; ++j) {
    int cj = tn + 16 * j + l15;                   // always < Npad
    wrow[j] = WT + (size_t)cj * Kpad;             // 16B-aligned rows
    float bv = 0.f;
    if (bias != nullptr) {                        // clamp + select: branch-free
      int bc = (cj < N) ? cj : (N - 1);
      bv = bias[bc];
      bv = (cj < N) ? bv : 0.f;
    }
#pragma unroll
    for (int r = 0; r < 8; ++r) acc[j][r] = bv;
  }

  const int ar = rowok ? arow : (M - 1);
  const float* arowp = A + (size_t)ar * K;

  for (int k0 = 0; k0 < K; k0 += 32) {
    // ---- all 8 B-fragment b128 loads issued up-front (one clause) ----
    v8h b0[4], b1[4];
#pragma unroll
    for (int j = 0; j < 4; ++j) {
      b0[j] = *(const v8h*)(wrow[j] + k0 + 8 * half);
      b1[j] = *(const v8h*)(wrow[j] + k0 + 16 + 8 * half);
    }
    // ---- A fragment (loaded once, reused by 4 WMMAs) ----
    v16h af;
    if (tileMf && (k0 + 32 <= K)) {
      const float* p0 = arowp + k0 + 8 * half;    // fast path: plain loads
      const float* p1 = p0 + 16;
#pragma unroll
      for (int i = 0; i < 8; ++i) {
        af[i]     = (_Float16)p0[i];
        af[i + 8] = (_Float16)p1[i];
      }
    } else {                                      // edge: clamp + select
#pragma unroll
      for (int i = 0; i < 16; ++i) {
        int kg = k0 + i + 8 * half + ((i >= 8) ? 8 : 0);
        int kc = (kg < K) ? kg : (K - 1);
        float av = arowp[kc];
        af[i] = (_Float16)((rowok && kg < K) ? av : 0.f);
      }
    }
    // ---- 4 WMMAs drain the staged fragments ----
#pragma unroll
    for (int j = 0; j < 4; ++j) {
      v16h bf;
#pragma unroll
      for (int i = 0; i < 8; ++i) { bf[i] = b0[j][i]; bf[i + 8] = b1[j][i]; }
      acc[j] = __builtin_amdgcn_wmma_f32_16x16x32_f16(false, af, false, bf,
                                                      (short)0, acc[j], false, false);
    }
  }
#pragma unroll
  for (int j = 0; j < 4; ++j) {
    int cj = tn + 16 * j + l15;
#pragma unroll
    for (int r = 0; r < 8; ++r) {
      int mrow = tm + r + 8 * half;
      if (mrow < M && cj < N) {
        float v = acc[j][r];
        if (relu) v = fmaxf(v, 0.f);
        C[(size_t)mrow * N + cj] = v;
      }
    }
  }
}

// ---------------- [B,3,N] -> [B,N,3] transpose ------------------------------
__global__ void k_transpose(const float* __restrict__ in, float* __restrict__ out,
                            int B, int N)
{
  size_t total = (size_t)B * N * 3;
  GRID_STRIDE(t, total) {
    int c = (int)(t % 3);
    size_t r = t / 3;
    int n = (int)(r % N);
    int b = (int)(r / N);
    out[t] = in[((size_t)b * 3 + c) * N + n];
  }
}

// ---------------- Farthest point sampling (xyz + dist fully in LDS) ---------
// One block per batch. LDS: xyz[3N] + dist[N] + reduction = 16N + 2KB bytes
// (258KB for N=16384 -- fits CDNA5's 320KB WGP LDS). The 512 serial
// iterations run entirely out of LDS after the initial stage-in.
__global__ void k_fps(const float* __restrict__ xyz, float* __restrict__ centers,
                      int B, int N, int m)
{
  extern __shared__ char smem_[];
  float* sx   = (float*)smem_;                 // 3*N  (AoS xyz)
  float* dist = sx + (size_t)3 * N;            // N
  float* rv   = dist + N;                      // blockDim
  int*   ri   = (int*)(rv + blockDim.x);       // blockDim
  int b = blockIdx.x;
  const float* P = xyz + (size_t)b * N * 3;
  float* Cc = centers + (size_t)b * m * 3;
  int tid = threadIdx.x, nt = blockDim.x;
  for (int j = tid; j < 3 * N; j += nt) sx[j] = P[j];
  for (int j = tid; j < N; j += nt) dist[j] = 1e10f;
  __syncthreads();
  int far = 0;
  for (int it = 0; it < m; ++it) {
    float px = sx[far * 3], py = sx[far * 3 + 1], pz = sx[far * 3 + 2];
    if (tid == 0) { Cc[it * 3] = px; Cc[it * 3 + 1] = py; Cc[it * 3 + 2] = pz; }
    float bv = -1.f; int bi = 0;
    for (int j = tid; j < N; j += nt) {
      float dx = sx[j * 3] - px, dy = sx[j * 3 + 1] - py, dz = sx[j * 3 + 2] - pz;
      float d = dx * dx + dy * dy + dz * dz;
      d = fminf(dist[j], d);
      dist[j] = d;
      if (d > bv) { bv = d; bi = j; }
    }
    rv[tid] = bv; ri[tid] = bi;
    __syncthreads();
    for (int s = nt >> 1; s > 0; s >>= 1) {
      if (tid < s) {
        float ov = rv[tid + s]; int oi = ri[tid + s];
        if (ov > rv[tid] || (ov == rv[tid] && oi < ri[tid])) { rv[tid] = ov; ri[tid] = oi; }
      }
      __syncthreads();
    }
    far = ri[0];
    __syncthreads();
  }
}

// ---------------- Brute-force kNN (dist array in LDS, k argmin rounds) ------
__global__ void k_knn(const float* __restrict__ q, const float* __restrict__ ref,
                      int* __restrict__ idx, float* __restrict__ d2out,
                      int B, int Mq, int Nr, int K)
{
  extern __shared__ char smem_[];
  float* dist = (float*)smem_;                 // Nr
  float* rv   = dist + Nr;                     // blockDim
  int*   ri   = (int*)(rv + blockDim.x);
  int gq = blockIdx.x;                         // b*Mq + mq
  int b = gq / Mq;
  const float* Q = q + (size_t)gq * 3;
  const float* R = ref + (size_t)b * Nr * 3;
  int tid = threadIdx.x, nt = blockDim.x;
  float qx = Q[0], qy = Q[1], qz = Q[2];
  for (int j = tid; j < Nr; j += nt) {
    float dx = R[j * 3] - qx, dy = R[j * 3 + 1] - qy, dz = R[j * 3 + 2] - qz;
    dist[j] = dx * dx + dy * dy + dz * dz;
  }
  __syncthreads();
  for (int kk = 0; kk < K; ++kk) {
    float bv = 3.4e38f; int bi = 0;
    for (int j = tid; j < Nr; j += nt) {
      float d = dist[j];
      if (d < bv) { bv = d; bi = j; }
    }
    rv[tid] = bv; ri[tid] = bi;
    __syncthreads();
    for (int s = nt >> 1; s > 0; s >>= 1) {
      if (tid < s) {
        float ov = rv[tid + s]; int oi = ri[tid + s];
        if (ov < rv[tid] || (ov == rv[tid] && oi < ri[tid])) { rv[tid] = ov; ri[tid] = oi; }
      }
      __syncthreads();
    }
    if (tid == 0) {
      idx[(size_t)gq * K + kk] = ri[0];
      if (d2out) d2out[(size_t)gq * K + kk] = fmaxf(rv[0], 0.f);
      dist[ri[0]] = 3.4e38f;
    }
    __syncthreads();
  }
}

// ---------------- SA edge feature build:  [gather(feats), gather(xyz)-center]
__global__ void k_edge(const float* __restrict__ feats, const float* __restrict__ xyz,
                       const float* __restrict__ cent, const int* __restrict__ idx,
                       float* __restrict__ edge, int B, int Nr, int Mq, int K, int Cf)
{
  int Ce = Cf + 3;
  size_t total = (size_t)B * Mq * K * Ce;
  GRID_STRIDE(t, total) {
    int c = (int)(t % Ce);
    size_t r = t / Ce;
    size_t bm = r / K;
    int mq = (int)(bm % Mq);
    int b = (int)(bm / Mq);
    int id = idx[r];
    float v;
    if (c < Cf) v = feats[((size_t)b * Nr + id) * Cf + c];
    else {
      int a = c - Cf;
      v = xyz[((size_t)b * Nr + id) * 3 + a] - cent[((size_t)b * Mq + mq) * 3 + a];
    }
    edge[t] = v;
  }
}

// ---------------- max-pool over neighbor dim --------------------------------
__global__ void k_maxpool(const float* __restrict__ in, float* __restrict__ out,
                          int R, int K, int C)
{
  size_t total = (size_t)R * C;
  GRID_STRIDE(t, total) {
    int c = (int)(t % C);
    size_t r = t / C;
    float m = -3.4e38f;
    for (int j = 0; j < K; ++j) m = fmaxf(m, in[(r * K + j) * (size_t)C + c]);
    out[t] = m;
  }
}

// ---------------- rel-pos + dt rows for the LPT positional MLP --------------
__global__ void k_reldt(const float* __restrict__ xl, const float* __restrict__ xi,
                        const int* __restrict__ idx, float dt,
                        float* __restrict__ rel, int B, int M, int K)
{
  size_t total = (size_t)B * M * K;
  GRID_STRIDE(t, total) {
    size_t bm = t / K;
    int b = (int)(bm / M);
    int id = idx[t];
    const float* a = xl + bm * 3;
    const float* r = xi + ((size_t)b * M + id) * 3;
    rel[t * 4 + 0] = a[0] - r[0];
    rel[t * 4 + 1] = a[1] - r[1];
    rel[t * 4 + 2] = a[2] - r[2];
    rel[t * 4 + 3] = dt;
  }
}

// ---------------- fused local point-transformer attention -------------------
__global__ void k_lpt_attn(const float* __restrict__ q, const float* __restrict__ kb,
                           const float* __restrict__ vb, const float* __restrict__ pe,
                           const int* __restrict__ idx, float* __restrict__ out,
                           int B, int M, int K, int C)
{
  extern __shared__ char smem_[];
  float* qs  = (float*)smem_;        // C
  float* lg  = qs + C;               // K
  float* red = lg + K;               // blockDim
  int bm = blockIdx.x;
  int b = bm / M;
  int tid = threadIdx.x, nt = blockDim.x;
  for (int c = tid; c < C; c += nt) qs[c] = q[(size_t)bm * C + c];
  __syncthreads();
  float scale = rsqrtf((float)C);
  for (int j = 0; j < K; ++j) {
    int id = idx[(size_t)bm * K + j];
    const float* kr = kb + ((size_t)b * M + id) * C;
    const float* pr = pe + ((size_t)bm * K + j) * C;
    float p = 0.f;
    for (int c = tid; c < C; c += nt) p += qs[c] * (kr[c] + pr[c]);
    red[tid] = p;
    __syncthreads();
    for (int s = nt >> 1; s > 0; s >>= 1) {
      if (tid < s) red[tid] += red[tid + s];
      __syncthreads();
    }
    if (tid == 0) lg[j] = red[0] * scale;
    __syncthreads();
  }
  if (tid == 0) {
    float mx = -3.4e38f;
    for (int j = 0; j < K; ++j) mx = fmaxf(mx, lg[j]);
    float s = 0.f;
    for (int j = 0; j < K; ++j) { float e = __expf(lg[j] - mx); lg[j] = e; s += e; }
    float inv = 1.f / s;
    for (int j = 0; j < K; ++j) lg[j] *= inv;
  }
  __syncthreads();
  for (int c = tid; c < C; c += nt) {
    float acc = 0.f;
    for (int j = 0; j < K; ++j) {
      int id = idx[(size_t)bm * K + j];
      acc += lg[j] * (vb[((size_t)b * M + id) * C + c] + pe[((size_t)bm * K + j) * C + c]);
    }
    out[(size_t)bm * C + c] = acc;
  }
}

// ---------------- fused temporal MHA core (Tm history steps, H heads) -------
__global__ void k_mha_attn(const float* __restrict__ Q, const float* __restrict__ Kt,
                           const float* __restrict__ Vt, float* __restrict__ out,
                           int P, int Tm, int C, int H)
{
  extern __shared__ char smem_[];
  float* buf = (float*)smem_;    // C
  float* lg  = buf + C;          // H*Tm
  int p = blockIdx.x;
  int tid = threadIdx.x;
  int Dh = C / H;
  float scale = rsqrtf((float)Dh);
  for (int t = 0; t < Tm; ++t) {
    buf[tid] = Q[(size_t)p * C + tid] * Kt[((size_t)t * P + p) * C + tid];
    __syncthreads();
    if (tid < H) {
      float s = 0.f;
      for (int d = 0; d < Dh; ++d) s += buf[tid * Dh + d];
      lg[tid * Tm + t] = s * scale;
    }
    __syncthreads();
  }
  if (tid < H) {
    float mx = -3.4e38f;
    for (int t = 0; t < Tm; ++t) mx = fmaxf(mx, lg[tid * Tm + t]);
    float s = 0.f;
    for (int t = 0; t < Tm; ++t) { float e = __expf(lg[tid * Tm + t] - mx); lg[tid * Tm + t] = e; s += e; }
    float inv = 1.f / s;
    for (int t = 0; t < Tm; ++t) lg[tid * Tm + t] *= inv;
  }
  __syncthreads();
  int h = tid / Dh;
  float acc = 0.f;
  for (int t = 0; t < Tm; ++t) acc += lg[h * Tm + t] * Vt[((size_t)t * P + p) * C + tid];
  out[(size_t)p * C + tid] = acc;
}

// ---------------- row-wise concat -------------------------------------------
__global__ void k_concat(const float* __restrict__ A, const float* __restrict__ Bm,
                         float* __restrict__ out, int R, int Ca, int Cb)
{
  int Ct = Ca + Cb;
  size_t total = (size_t)R * Ct;
  GRID_STRIDE(t, total) {
    int c = (int)(t % Ct);
    size_t r = t / Ct;
    out[t] = (c < Ca) ? A[r * Ca + c] : Bm[r * Cb + (c - Ca)];
  }
}

// ---------------- inverse-sq-distance weighted interpolation ----------------
__global__ void k_fpinterp(const float* __restrict__ xsrc, const int* __restrict__ idx,
                           const float* __restrict__ d2, float* __restrict__ out,
                           int B, int Ms, int Mt, int K, int C)
{
  extern __shared__ char smem_[];
  float* w = (float*)smem_;    // K
  int bm = blockIdx.x;
  int b = bm / Mt;
  int tid = threadIdx.x, nt = blockDim.x;
  if (tid == 0) {
    float s = 0.f;
    for (int j = 0; j < K; ++j) { float ww = 1.f / (d2[(size_t)bm * K + j] + 1e-8f); w[j] = ww; s += ww; }
    float inv = 1.f / s;
    for (int j = 0; j < K; ++j) w[j] *= inv;
  }
  __syncthreads();
  for (int c = tid; c < C; c += nt) {
    float acc = 0.f;
    for (int j = 0; j < K; ++j)
      acc += w[j] * xsrc[((size_t)b * Ms + idx[(size_t)bm * K + j]) * C + c];
    out[(size_t)bm * C + c] = acc;
  }
}

// ---------------- residual add ----------------------------------------------
__global__ void k_add(const float* __restrict__ a, const float* __restrict__ b,
                      float* __restrict__ o, size_t n)
{
  GRID_STRIDE(t, n) o[t] = a[t] + b[t];
}

// ---------------- host-side helpers -----------------------------------------
struct Arena {
  char* base; size_t off; size_t cap;
  void* alloc(size_t bytes) {
    size_t a = (off + 255) & ~(size_t)255;
    off = a + bytes;
    return base + a;
  }
  float*    f(size_t n) { return (float*)alloc(n * 4); }
  int*      i(size_t n) { return (int*)alloc(n * 4); }
  _Float16* h(size_t n) { return (_Float16*)alloc(n * 2); }
};

static inline unsigned gblocks(size_t total, int thr) {
  size_t b = (total + (size_t)thr - 1) / (size_t)thr;
  if (b > (size_t)1048576) b = 1048576;
  if (b == 0) b = 1;
  return (unsigned)b;
}

// Parameter flattening (setup_inputs insertion order, tuples/lists in order):
enum {
  SA1 = 0,   // (W,b)x3 : 6->32->32->64
  SA2 = 6,   // 67->96->96->128
  SA3 = 12,  // 131->128->128->256
  LPT1 = 18, LPT2 = 23, LPT3 = 28,   // Wq,Wk,Wv,Wpos,bpos
  MHA1 = 33, MHA2 = 41, MHA3 = 49,   // Wq,Wk,Wv,Wo,bq,bk,bv,bo
  FP32 = 57, // (W,b)x3 : 768->512->256->256
  FP21 = 63, // (W,b)x2 : 384->256->256
  FP10 = 67, // (W,b)x2 : 256->256->256
  CLS = 71,  // W x3 : 256->128->64->3
  IN_XYZ = 74, IN_NUMPRED = 75
};

extern "C" void kernel_launch(void* const* d_in, const int* in_sizes, int n_in,
                              void* d_out, int out_size, void* d_ws, size_t ws_size,
                              hipStream_t stream)
{
  (void)in_sizes; (void)n_in; (void)out_size;
  const int B = 4, N = 16384;
  const int M1 = 512, M2 = 256, M3 = 128;
  const int K1 = 32, K2 = 16, K3 = 8;
  const int C1 = 64, C2 = 128, C3 = 256;
  const int H = 4;

  const float* pp[74];
  for (int i = 0; i < 74; ++i) pp[i] = (const float*)d_in[i];
  const float* XYZ = (const float*)d_in[IN_XYZ];
  float* out = (float*)d_out;

  Arena ws{(char*)d_ws, 0, ws_size};

  // ---- pre-pack every weight matrix to f16 [Npad%64][Kpad%32] (persistent) --
  const _Float16* pw[74] = {nullptr};
  int kpad[74] = {0};
  struct WSpec { int idx, K, N; };
  const WSpec wspecs[] = {
    {SA1 + 0, 6, 32},   {SA1 + 2, 32, 32},  {SA1 + 4, 32, 64},
    {SA2 + 0, 67, 96},  {SA2 + 2, 96, 96},  {SA2 + 4, 96, 128},
    {SA3 + 0, 131, 128},{SA3 + 2, 128, 128},{SA3 + 4, 128, 256},
    {LPT1 + 0, 64, 64}, {LPT1 + 1, 64, 64}, {LPT1 + 2, 64, 64}, {LPT1 + 3, 4, 64},
    {LPT2 + 0, 128, 128},{LPT2 + 1, 128, 128},{LPT2 + 2, 128, 128},{LPT2 + 3, 4, 128},
    {LPT3 + 0, 256, 256},{LPT3 + 1, 256, 256},{LPT3 + 2, 256, 256},{LPT3 + 3, 4, 256},
    {MHA1 + 0, 64, 64}, {MHA1 + 1, 64, 64}, {MHA1 + 2, 64, 64}, {MHA1 + 3, 64, 64},
    {MHA2 + 0, 128, 128},{MHA2 + 1, 128, 128},{MHA2 + 2, 128, 128},{MHA2 + 3, 128, 128},
    {MHA3 + 0, 256, 256},{MHA3 + 1, 256, 256},{MHA3 + 2, 256, 256},{MHA3 + 3, 256, 256},
    {FP32 + 0, 768, 512},{FP32 + 2, 512, 256},{FP32 + 4, 256, 256},
    {FP21 + 0, 384, 256},{FP21 + 2, 256, 256},
    {FP10 + 0, 256, 256},{FP10 + 2, 256, 256},
    {CLS + 0, 256, 128}, {CLS + 1, 128, 64},  {CLS + 2, 64, 3},
  };
  for (const WSpec& s : wspecs) {
    int Kp = (s.K + 31) & ~31;
    int Np = (s.N + 63) & ~63;          // pad cols to 64 for the 16x64 tiles
    _Float16* dst = ws.h((size_t)Kp * Np);
    k_packw<<<gblocks((size_t)Kp * Np, 256), 256, 0, stream>>>(pp[s.idx], dst, s.K, s.N, Kp, Np);
    pw[s.idx] = dst;
    kpad[s.idx] = Kp;
  }

  // ---- persistent feature buffers ----
  float* frame0 = ws.f((size_t)B * N * 3);
  float* frame1 = ws.f((size_t)B * N * 3);
  float *f1s[3], *f2s[3], *f3s[3], *x1s[3], *x2s[3], *x3s[3];
  for (int s = 0; s < 3; ++s) {
    f1s[s] = ws.f((size_t)B * M1 * C1); x1s[s] = ws.f((size_t)B * M1 * 3);
    f2s[s] = ws.f((size_t)B * M2 * C2); x2s[s] = ws.f((size_t)B * M2 * 3);
    f3s[s] = ws.f((size_t)B * M3 * C3); x3s[s] = ws.f((size_t)B * M3 * 3);
  }

  auto gemm = [&](const float* A, int widx, const float* bias, float* Cm,
                  int M, int K, int Nn, int relu) {
    int tiles = ((M + 15) / 16) * ((Nn + 63) / 64);
    int blocks = (tiles + 7) / 8;                       // 8 waves / block
    k_gemm<<<blocks, 256, 0, stream>>>(A, pw[widx], bias, Cm, M, K, Nn, kpad[widx], relu);
  };

  // ---- set abstraction level: FPS -> kNN -> edge -> MLP(WMMA) -> maxpool ----
  auto sa_level = [&](const float* feats, const float* xyz, int Nr, int Cf,
                      int Mq, int Kk, int pbase, int h0, int h1, int h2,
                      float* fout, float* cent) {
    size_t mk = ws.off;
    k_fps<<<B, 256, (size_t)Nr * 16 + 256 * 8, stream>>>(xyz, cent, B, Nr, Mq);
    int* idx = ws.i((size_t)B * Mq * Kk);
    k_knn<<<B * Mq, 128, (size_t)Nr * 4 + 128 * 8, stream>>>(cent, xyz, idx, nullptr, B, Mq, Nr, Kk);
    int Ce = Cf + 3;
    size_t Rows = (size_t)B * Mq * Kk;
    float* e0 = ws.f(Rows * Ce);
    k_edge<<<gblocks(Rows * Ce, 256), 256, 0, stream>>>(feats, xyz, cent, idx, e0, B, Nr, Mq, Kk, Cf);
    float* a1 = ws.f(Rows * h0); gemm(e0, pbase + 0, pp[pbase + 1], a1, (int)Rows, Ce, h0, 1);
    float* a2 = ws.f(Rows * h1); gemm(a1, pbase + 2, pp[pbase + 3], a2, (int)Rows, h0, h1, 1);
    float* a3 = ws.f(Rows * h2); gemm(a2, pbase + 4, pp[pbase + 5], a3, (int)Rows, h1, h2, 0);
    k_maxpool<<<gblocks((size_t)B * Mq * h2, 256), 256, 0, stream>>>(a3, fout, B * Mq, Kk, h2);
    ws.off = mk;
  };

  auto encode = [&](const float* pc, int s) {
    sa_level(pc,     pc,     N,  3,  M1, K1, SA1, 32,  32,  64,  f1s[s], x1s[s]);
    sa_level(f1s[s], x1s[s], M1, C1, M2, K2, SA2, 96,  96,  128, f2s[s], x2s[s]);
    sa_level(f2s[s], x2s[s], M2, C2, M3, K3, SA3, 128, 128, 256, f3s[s], x3s[s]);
  };

  // ---- local point-transformer attention for one (level, frame) pair ----
  auto lpt_level = [&](int pbase, int M, int C, int Kk, const float* qb,
                       const float* f_i, const float* x_i, const float* x_last,
                       float dt, float* ftout) {
    size_t mk = ws.off;
    int P = B * M;
    float* kb = ws.f((size_t)P * C); gemm(f_i, pbase + 1, nullptr, kb, P, C, C, 0);
    float* vb = ws.f((size_t)P * C); gemm(f_i, pbase + 2, nullptr, vb, P, C, C, 0);
    int* idx = ws.i((size_t)P * Kk);
    k_knn<<<P, 128, (size_t)M * 4 + 128 * 8, stream>>>(x_last, x_i, idx, nullptr, B, M, M, Kk);
    size_t Rk = (size_t)P * Kk;
    float* rel = ws.f(Rk * 4);
    k_reldt<<<gblocks(Rk, 256), 256, 0, stream>>>(x_last, x_i, idx, dt, rel, B, M, Kk);
    float* pe = ws.f(Rk * C);
    gemm(rel, pbase + 3, pp[pbase + 4], pe, (int)Rk, 4, C, 0);
    k_lpt_attn<<<P, 128, (size_t)(C + Kk + 128) * 4, stream>>>(qb, kb, vb, pe, idx, ftout, B, M, Kk, C);
    ws.off = mk;
  };

  // ---- temporal MHA + concat with last features ----
  auto mha_level = [&](int pbase, const float* ft, int Tm, int P, int C,
                       const float* flast, float* tmout) {
    size_t mk = ws.off;
    float* Qp = ws.f((size_t)P * C);
    gemm(ft + (size_t)(Tm - 1) * P * C, pbase + 0, pp[pbase + 4], Qp, P, C, C, 0);
    float* Kp = ws.f((size_t)Tm * P * C); gemm(ft, pbase + 1, pp[pbase + 5], Kp, Tm * P, C, C, 0);
    float* Vp = ws.f((size_t)Tm * P * C); gemm(ft, pbase + 2, pp[pbase + 6], Vp, Tm * P, C, C, 0);
    float* ao = ws.f((size_t)P * C);
    k_mha_attn<<<P, C, (size_t)(C + H * Tm) * 4, stream>>>(Qp, Kp, Vp, ao, P, Tm, C, H);
    float* pr = ws.f((size_t)P * C); gemm(ao, pbase + 3, pp[pbase + 7], pr, P, C, C, 0);
    k_concat<<<gblocks((size_t)P * 2 * C, 256), 256, 0, stream>>>(pr, flast, tmout, P, C, C);
    ws.off = mk;
  };

  // ---- full prediction step ----
  auto pred_next = [&](const float* last0, int T, float* pred) {
    size_t mk0 = ws.off;
    int Tm = T - 1;
    float* ft1 = ws.f((size_t)Tm * B * M1 * C1);
    float* ft2 = ws.f((size_t)Tm * B * M2 * C2);
    float* ft3 = ws.f((size_t)Tm * B * M3 * C3);
    float* q1 = ws.f((size_t)B * M1 * C1);
    float* q2 = ws.f((size_t)B * M2 * C2);
    float* q3 = ws.f((size_t)B * M3 * C3);
    gemm(f1s[T - 1], LPT1 + 0, nullptr, q1, B * M1, C1, C1, 0);
    gemm(f2s[T - 1], LPT2 + 0, nullptr, q2, B * M2, C2, C2, 0);
    gemm(f3s[T - 1], LPT3 + 0, nullptr, q3, B * M3, C3, C3, 0);
    for (int i = 0; i < Tm; ++i) {
      float dt = 1.f - (float)i / (float)T;
      lpt_level(LPT1, M1, C1, K1, q1, f1s[i], x1s[i], x1s[T - 1], dt, ft1 + (size_t)i * B * M1 * C1);
      lpt_level(LPT2, M2, C2, K2, q2, f2s[i], x2s[i], x2s[T - 1], dt, ft2 + (size_t)i * B * M2 * C2);
      lpt_level(LPT3, M3, C3, K3, q3, f3s[i], x3s[i], x3s[T - 1], dt, ft3 + (size_t)i * B * M3 * C3);
    }
    float* tmo1 = ws.f((size_t)B * M1 * 2 * C1);
    float* tmo2 = ws.f((size_t)B * M2 * 2 * C2);
    float* tmo3 = ws.f((size_t)B * M3 * 2 * C3);
    mha_level(MHA1, ft1, Tm, B * M1, C1, f1s[T - 1], tmo1);
    mha_level(MHA2, ft2, Tm, B * M2, C2, f2s[T - 1], tmo2);
    mha_level(MHA3, ft3, Tm, B * M3, C3, f3s[T - 1], tmo3);

    float* x2o = ws.f((size_t)B * M2 * 256);
    float* x1o = ws.f((size_t)B * M1 * 256);
    float* x0  = ws.f((size_t)B * N * 256);
    { // fp32: tm3 -> x2_last grid (skip tm2), 768->512->256->256
      size_t mk = ws.off; int P = B * M2;
      int* idx = ws.i((size_t)P * K3); float* d2 = ws.f((size_t)P * K3);
      k_knn<<<P, 128, (size_t)M3 * 4 + 128 * 8, stream>>>(x2s[T - 1], x3s[T - 1], idx, d2, B, M2, M3, K3);
      float* feat = ws.f((size_t)P * 512);
      k_fpinterp<<<P, 256, (size_t)K3 * 4, stream>>>(tmo3, idx, d2, feat, B, M3, M2, K3, 512);
      float* cat = ws.f((size_t)P * 768);
      k_concat<<<gblocks((size_t)P * 768, 256), 256, 0, stream>>>(feat, tmo2, cat, P, 512, 256);
      float* h1 = ws.f((size_t)P * 512); gemm(cat, FP32 + 0, pp[FP32 + 1], h1, P, 768, 512, 1);
      float* h2 = ws.f((size_t)P * 256); gemm(h1, FP32 + 2, pp[FP32 + 3], h2, P, 512, 256, 1);
      gemm(h2, FP32 + 4, pp[FP32 + 5], x2o, P, 256, 256, 0);
      ws.off = mk;
    }
    { // fp21: x2o -> x1_last grid (skip tm1), 384->256->256
      size_t mk = ws.off; int P = B * M1;
      int* idx = ws.i((size_t)P * K2); float* d2 = ws.f((size_t)P * K2);
      k_knn<<<P, 128, (size_t)M2 * 4 + 128 * 8, stream>>>(x1s[T - 1], x2s[T - 1], idx, d2, B, M1, M2, K2);
      float* feat = ws.f((size_t)P * 256);
      k_fpinterp<<<P, 256, (size_t)K2 * 4, stream>>>(x2o, idx, d2, feat, B, M2, M1, K2, 256);
      float* cat = ws.f((size_t)P * 384);
      k_concat<<<gblocks((size_t)P * 384, 256), 256, 0, stream>>>(feat, tmo1, cat, P, 256, 128);
      float* h1 = ws.f((size_t)P * 256); gemm(cat, FP21 + 0, pp[FP21 + 1], h1, P, 384, 256, 1);
      gemm(h1, FP21 + 2, pp[FP21 + 3], x1o, P, 256, 256, 0);
      ws.off = mk;
    }
    { // fp10: x1o -> full point grid (no skip), 256->256->256
      size_t mk = ws.off; int P = B * N;
      int* idx = ws.i((size_t)P * K1); float* d2 = ws.f((size_t)P * K1);
      k_knn<<<P, 128, (size_t)M1 * 4 + 128 * 8, stream>>>(last0, x1s[T - 1], idx, d2, B, N, M1, K1);
      float* feat = ws.f((size_t)P * 256);
      k_fpinterp<<<P, 256, (size_t)K1 * 4, stream>>>(x1o, idx, d2, feat, B, M1, N, K1, 256);
      float* h1 = ws.f((size_t)P * 256); gemm(feat, FP10 + 0, pp[FP10 + 1], h1, P, 256, 256, 1);
      gemm(h1, FP10 + 2, pp[FP10 + 3], x0, P, 256, 256, 0);
      ws.off = mk;
    }
    { // cls head + residual
      size_t mk = ws.off; int P = B * N;
      float* c1 = ws.f((size_t)P * 128); gemm(x0, CLS + 0, nullptr, c1, P, 256, 128, 0);
      float* c2 = ws.f((size_t)P * 64);  gemm(c1, CLS + 1, nullptr, c2, P, 128, 64, 0);
      float* dl = ws.f((size_t)P * 3);   gemm(c2, CLS + 2, nullptr, dl, P, 64, 3, 0);
      k_add<<<gblocks((size_t)P * 3, 256), 256, 0, stream>>>(last0, dl, pred, (size_t)P * 3);
      ws.off = mk;
    }
    ws.off = mk0;
  };

  // ---- pipeline: 2 input frames, 2 autoregressive predictions ----
  k_transpose<<<gblocks((size_t)B * N * 3, 256), 256, 0, stream>>>(XYZ, frame0, B, N);
  k_transpose<<<gblocks((size_t)B * N * 3, 256), 256, 0, stream>>>(XYZ + (size_t)B * 3 * N, frame1, B, N);

  encode(frame0, 0);
  encode(frame1, 1);

  float* pred0 = out;                        // [B,N,3]
  float* pred1 = out + (size_t)B * N * 3;    // [B,N,3]

  pred_next(frame1, 2, pred0);   // T = 2 history frames
  encode(pred0, 2);
  pred_next(pred0, 3, pred1);    // T = 3 history frames
}
